// Indexer_42571715838077
// MI455X (gfx1250) — compile-verified, matrix operand order checked by reading, not run on hardware
//
#include <hip/hip_runtime.h>
#include <hip/hip_bf16.h>

#define S_SEQ 2048
#define HID 7168
#define QLORA 1536
#define NH 32
#define HD 128
#define TOPK_N 512
#define NEGV -1000000000.0f

typedef __attribute__((ext_vector_type(16))) __bf16 v16bf;
typedef __attribute__((ext_vector_type(8)))  __bf16 v8bf;
typedef __attribute__((ext_vector_type(8)))  float  v8f;

// Load one 16x32 bf16 WMMA fragment (A-style; B uses same pattern with N=r).
// p = row_base + k0 + 8*half ; covers K {0..7,16..23} (half=0) / {8..15,24..31} (half=1)
__device__ __forceinline__ v16bf load_frag(const __bf16* p) {
  v16bf f;
  ((v8bf*)&f)[0] = *(const v8bf*)(p);        // VGPR0-3
  ((v8bf*)&f)[1] = *(const v8bf*)(p + 16);   // VGPR4-7
  return f;
}

// ---- gfx1250 async copy global->LDS (ASYNCcnt-tracked), per-lane 16B ----
// Generic LDS pointer low 32 bits == wave-relative LDS byte offset.
__device__ __forceinline__ unsigned lds_off(const void* p) {
  return (unsigned)(unsigned long long)p;
}
__device__ __forceinline__ void async_copy_b128(unsigned lds_addr, unsigned long long gaddr) {
  asm volatile("global_load_async_to_lds_b128 %0, %1, off"
               :: "v"(lds_addr), "v"(gaddr) : "memory");
}
__device__ __forceinline__ void async_wait0() {
  asm volatile("s_wait_asynccnt 0x0" ::: "memory");
}

// ---------------- prep: f32 -> bf16 conversions ----------------
__global__ void convert_qr_kernel(const float* __restrict__ qr, __bf16* __restrict__ qr_bf, int n) {
  int i = blockIdx.x * 256 + threadIdx.x;
  if (i < n) qr_bf[i] = (__bf16)qr[i];
}

__global__ void transpose_wqb_kernel(const float* __restrict__ wqb, __bf16* __restrict__ wqbT) {
  int i = blockIdx.x * 256 + threadIdx.x;           // over 4096*1536
  if (i < QLORA * NH * HD) {
    int nn = i / QLORA;                             // output row   (0..4095)
    int kk = i - nn * QLORA;                        // output col   (0..1535)
    wqbT[i] = (__bf16)wqb[(size_t)kk * (NH * HD) + nn];
  }
}

// ---------------- k = rmsnorm(x@wk) + rope ; weights = x@ww * 1/64 ----------------
__global__ __launch_bounds__(128) void k_weights_kernel(
    const float* __restrict__ x, const float* __restrict__ wk,
    const float* __restrict__ knw, const float* __restrict__ ww,
    const float* __restrict__ fcos, const float* __restrict__ fsin,
    __bf16* __restrict__ kbf, float* __restrict__ weights)
{
  int t = blockIdx.x, d = threadIdx.x;
  const float* xr = x + (size_t)t * HID;
  __shared__ float xs[128];
  __shared__ float kv[128];
  __shared__ float red[128];
  float acc = 0.f;
  for (int i0 = 0; i0 < HID; i0 += 128) {
    __syncthreads();
    xs[d] = xr[i0 + d];
    __syncthreads();
    #pragma unroll 8
    for (int j = 0; j < 128; ++j) acc = fmaf(xs[j], wk[(size_t)(i0 + j) * HD + d], acc);
  }
  // RMS norm over 128
  red[d] = acc * acc; __syncthreads();
  for (int sft = 64; sft > 0; sft >>= 1) { if (d < sft) red[d] += red[d + sft]; __syncthreads(); }
  float scale = rsqrtf(red[0] * (1.0f / HD) + 1e-6f);
  float kn = acc * scale * knw[d];
  kv[d] = kn; __syncthreads();
  float outv;
  if (d < 32)      { float c = fcos[t*32+d],      s = fsin[t*32+d];      outv = kv[d]    * c - kv[d+32] * s; }
  else if (d < 64) { int j = d-32; float c = fcos[t*32+j], s = fsin[t*32+j]; outv = kv[j] * s + kv[d] * c; }
  else outv = kn;
  kbf[(size_t)t * HD + d] = (__bf16)outv;
  // weights[t][h] = (x . ww[:,h]) / 64
  int h = d & 31, g = d >> 5;
  float wp = 0.f;
  for (int i = g; i < HID; i += 4) wp = fmaf(xr[i], ww[(size_t)i * NH + h], wp);
  __syncthreads();
  red[d] = wp; __syncthreads();
  if (g == 0) weights[t * NH + h] = (red[h] + red[32 + h] + red[64 + h] + red[96 + h]) * (1.0f / 64.0f);
}

// ---------------- q = qr @ wq_b (bf16 WMMA, LDS-staged A) + rope -> qbf[h][t][d] ----------------
__global__ __launch_bounds__(256) void q_gemm_rope_kernel(
    const __bf16* __restrict__ qr_bf, const __bf16* __restrict__ wqbT,
    const float* __restrict__ fcos, const float* __restrict__ fsin,
    __bf16* __restrict__ qbf)
{
  int bx = blockIdx.x;
  int h  = bx & (NH - 1);
  int t0 = (bx >> 5) * 16;
  int tid = threadIdx.x;
  int wave = tid >> 5, lane = tid & 31;
  int r = lane & 15, half = lane >> 4;
  int ncol = wave * 16;                              // col base within head (0..112)

  __shared__ __bf16 as_[16 * QLORA];                 // 48 KB: A tile rows t0..t0+15 (shared by 8 waves)
  __shared__ float  qt[16][HD];                      // 8 KB : f32 result tile for rope

  // async-stage A block: 16*1536*2B = 48KB = 3072 x b128 ; 12 chunks/thread
  {
    const __bf16* ga = qr_bf + (size_t)t0 * QLORA;
    #pragma unroll
    for (int c = 0; c < 12; ++c) {
      int e = c * 256 + tid;
      async_copy_b128(lds_off(&as_[e * 8]), (unsigned long long)(const void*)(ga + e * 8));
    }
  }
  async_wait0();
  __syncthreads();

  const __bf16* arow = &as_[r * QLORA + 8 * half];
  const __bf16* brow = wqbT + (size_t)(h * HD + ncol + r) * QLORA + 8 * half;
  v8f acc = {};
  for (int k0 = 0; k0 < QLORA; k0 += 32) {
    v16bf a = load_frag(arow + k0);
    v16bf b = load_frag(brow + k0);
    acc = __builtin_amdgcn_wmma_f32_16x16x32_bf16(false, a, false, b, (short)0, acc, false, false);
  }
  #pragma unroll
  for (int i = 0; i < 8; ++i) qt[i + 8 * half][ncol + r] = acc[i];   // C layout: M=i+8*half, N=r
  __syncthreads();
  // apply rope on d<64 (pairs j / j+32), convert to bf16, store [h][t][d]
  for (int rep = 0; rep < 8; ++rep) {
    int e = rep * 256 + tid;
    int m = e >> 7, dc = e & 127;
    float v;
    if (dc < 32)      { float c = fcos[(t0+m)*32+dc], s = fsin[(t0+m)*32+dc]; v = qt[m][dc]   * c - qt[m][dc+32] * s; }
    else if (dc < 64) { int j = dc-32; float c = fcos[(t0+m)*32+j], s = fsin[(t0+m)*32+j]; v = qt[m][j] * s + qt[m][dc] * c; }
    else v = qt[m][dc];
    qbf[((size_t)h * S_SEQ + (t0 + m)) * HD + dc] = (__bf16)v;
  }
}

// ---------------- score[t][s] = sum_h relu(q_h[t].k[s]) * w[t][h] + causal mask ----------------
// LDS: K block staged once (32KB), Q head tile double-buffered via async copies (2x4KB).
__global__ __launch_bounds__(256) void score_kernel(
    const __bf16* __restrict__ qbf, const __bf16* __restrict__ kbf,
    const float* __restrict__ weights, float* __restrict__ score)
{
  int bx = blockIdx.x;
  int t0 = (bx >> 4) * 16;
  int sg = bx & 15;
  int tid = threadIdx.x;
  int wave = tid >> 5, lane = tid & 31;
  int r = lane & 15, half = lane >> 4;
  int sblk = sg * 128;
  int s0 = sblk + wave * 16;

  __shared__ __bf16 ks[128 * HD];        // 32 KB : K rows sblk..sblk+127
  __shared__ __bf16 qs[2][16 * HD];      // 2x4 KB: Q tile of one head, double buffered
  __shared__ float  wlds[16 * NH];       // 2 KB

  for (int e = tid; e < 16 * NH; e += 256) wlds[e] = weights[(t0 + (e >> 5)) * NH + (e & 31)];

  // async-stage K block: 128*256B = 2048 x b128 ; 8 chunks/thread
  {
    const __bf16* gk = kbf + (size_t)sblk * HD;
    #pragma unroll
    for (int c = 0; c < 8; ++c) {
      int e = c * 256 + tid;
      async_copy_b128(lds_off(&ks[e * 8]), (unsigned long long)(const void*)(gk + e * 8));
    }
  }
  // async-stage Q tile for head 0 (16*256B = 256 x b128 ; 1 chunk/thread)
  {
    const __bf16* gq = qbf + (size_t)t0 * HD;      // h = 0
    async_copy_b128(lds_off(&qs[0][tid * 8]), (unsigned long long)(const void*)(gq + tid * 8));
  }
  async_wait0();
  __syncthreads();

  const __bf16* brow = &ks[(wave * 16 + r) * HD + 8 * half];
  v8f sacc = {};
  for (int h = 0; h < NH; ++h) {
    int cur = h & 1;
    if (h + 1 < NH) {                    // prefetch next head's Q tile into other buffer
      const __bf16* gq = qbf + ((size_t)(h + 1) * S_SEQ + t0) * HD;
      async_copy_b128(lds_off(&qs[cur ^ 1][tid * 8]), (unsigned long long)(const void*)(gq + tid * 8));
    }
    const __bf16* arow = &qs[cur][r * HD + 8 * half];
    v8f hacc = {};
    #pragma unroll
    for (int k0 = 0; k0 < HD; k0 += 32) {
      v16bf a = load_frag(arow + k0);
      v16bf b = load_frag(brow + k0);
      hacc = __builtin_amdgcn_wmma_f32_16x16x32_bf16(false, a, false, b, (short)0, hacc, false, false);
    }
    #pragma unroll
    for (int i = 0; i < 8; ++i) {
      float w = wlds[(i + 8 * half) * NH + h];
      sacc[i] += fmaxf(hacc[i], 0.f) * w;
    }
    async_wait0();                       // next head's tile landed
    __syncthreads();                     // all waves done with qs[cur] before it is overwritten
  }
  #pragma unroll
  for (int i = 0; i < 8; ++i) {
    int m = t0 + i + 8 * half;
    int n = s0 + r;
    float v = sacc[i] + (n <= m ? 0.f : NEGV);
    score[(size_t)m * S_SEQ + n] = v;
  }
}

// ---------------- exact top-512 per row via radix-select, deterministic tie-break ----------------
__global__ __launch_bounds__(256) void topk_kernel(const float* __restrict__ score,
                                                   float* __restrict__ out)
{
  int t = blockIdx.x;
  int tid = threadIdx.x;
  const float* row = score + (size_t)t * S_SEQ;
  int base = tid * 8;
  unsigned key[8];
  #pragma unroll
  for (int j = 0; j < 8; ++j) {
    unsigned k = __float_as_uint(row[base + j]);
    key[j] = (k & 0x80000000u) ? ~k : (k | 0x80000000u);   // order-preserving map
  }
  __shared__ unsigned red[256];
  unsigned prefix = 0, hm = 0;
  unsigned kth = TOPK_N;
  for (int bit = 31; bit >= 0; --bit) {
    unsigned bm = 1u << bit;
    unsigned test = prefix | bm;
    unsigned hmb = hm | bm;
    unsigned c = 0;
    #pragma unroll
    for (int j = 0; j < 8; ++j) c += ((key[j] & hmb) == test) ? 1u : 0u;
    red[tid] = c; __syncthreads();
    for (int sft = 128; sft > 0; sft >>= 1) { if (tid < sft) red[tid] += red[tid + sft]; __syncthreads(); }
    unsigned ctot = red[0];
    __syncthreads();
    if (ctot >= kth) prefix = test; else kth -= ctot;
    hm = hmb;
  }
  // prefix == TOPK-th largest key
  unsigned cg = 0, ceq = 0;
  #pragma unroll
  for (int j = 0; j < 8; ++j) { cg += key[j] > prefix; ceq += key[j] == prefix; }
  red[tid] = cg; __syncthreads();
  for (int sft = 128; sft > 0; sft >>= 1) { if (tid < sft) red[tid] += red[tid + sft]; __syncthreads(); }
  unsigned total_gt = red[0];
  __syncthreads();
  unsigned need_eq = TOPK_N - total_gt;
  // inclusive scan of eq-counts (index order) for tie ranking
  red[tid] = ceq; __syncthreads();
  for (int sft = 1; sft < 256; sft <<= 1) {
    unsigned v = red[tid];
    unsigned add = (tid >= sft) ? red[tid - sft] : 0u;
    __syncthreads();
    red[tid] = v + add;
    __syncthreads();
  }
  unsigned rank = red[tid] - ceq;                    // exclusive prefix
  const float NEGINF = __int_as_float(0xff800000);
  unsigned local = 0;
  #pragma unroll
  for (int j = 0; j < 8; ++j) {
    bool sel;
    if (key[j] > prefix) sel = true;
    else if (key[j] == prefix) { sel = (rank + local) < need_eq; local++; }
    else sel = false;
    out[(size_t)t * S_SEQ + base + j] = sel ? 0.0f : NEGINF;
  }
}

extern "C" void kernel_launch(void* const* d_in, const int* in_sizes, int n_in,
                              void* d_out, int out_size, void* d_ws, size_t ws_size,
                              hipStream_t stream)
{
  const float* x    = (const float*)d_in[0];
  const float* qr   = (const float*)d_in[1];
  const float* fcos = (const float*)d_in[2];
  const float* fsin = (const float*)d_in[3];
  // d_in[4] = mask (computed analytically)
  const float* wqb  = (const float*)d_in[5];
  const float* wk   = (const float*)d_in[6];
  const float* knw  = (const float*)d_in[7];
  const float* ww   = (const float*)d_in[8];
  float* out = (float*)d_out;

  char* ws = (char*)d_ws;
  size_t off = 0;
  __bf16* qr_bf   = (__bf16*)(ws + off); off += (size_t)S_SEQ * QLORA * 2;          // 6.29 MB
  __bf16* wqbT    = (__bf16*)(ws + off); off += (size_t)NH * HD * QLORA * 2;        // 12.58 MB
  __bf16* kbf     = (__bf16*)(ws + off); off += (size_t)S_SEQ * HD * 2;             // 0.52 MB
  __bf16* qbf     = (__bf16*)(ws + off); off += (size_t)NH * S_SEQ * HD * 2;        // 16.78 MB
  float*  weights = (float*)(ws + off);  off += (size_t)S_SEQ * NH * 4;             // 0.26 MB
  float*  score   = (float*)(ws + off);  off += (size_t)S_SEQ * S_SEQ * 4;          // 16.78 MB

  int nqr = S_SEQ * QLORA;
  convert_qr_kernel<<<(nqr + 255) / 256, 256, 0, stream>>>(qr, qr_bf, nqr);
  int nw = QLORA * NH * HD;
  transpose_wqb_kernel<<<(nw + 255) / 256, 256, 0, stream>>>(wqb, wqbT);
  k_weights_kernel<<<S_SEQ, 128, 0, stream>>>(x, wk, knw, ww, fcos, fsin, kbf, weights);
  q_gemm_rope_kernel<<<NH * (S_SEQ / 16), 256, 0, stream>>>(qr_bf, wqbT, fcos, fsin, qbf);
  score_kernel<<<(S_SEQ / 16) * (S_SEQ / 128), 256, 0, stream>>>(qbf, kbf, weights, score);
  topk_kernel<<<S_SEQ, 256, 0, stream>>>(score, out);
}
